// PWEMoE_962072675075
// MI455X (gfx1250) — matrix operand order repaired and checked by reading, not compiled, not run on hardware
//
#include <hip/hip_runtime.h>
#include <math.h>

// ---------------- problem constants ----------------
constexpr int Bn = 8, Sn = 2048, Hn = 1024, Fn = 4096, En = 8;

// ---------------- types ----------------
typedef __bf16 v16bf __attribute__((ext_vector_type(16)));
typedef float  v8f   __attribute__((ext_vector_type(8)));

struct alignas(16) U4 { unsigned int a, b, c, d; };
struct alignas(16) F4 { float x, y, z, w; };

union FragCvt { U4 u[2]; v16bf v; };

__device__ inline unsigned short f2bf(float f) {
  unsigned int u = __builtin_bit_cast(unsigned int, f);
  unsigned int r = (u + 0x7FFFu + ((u >> 16) & 1u)) >> 16;
  return (unsigned short)r;
}

__device__ inline float gelu_tanh(float x) {
  float x3 = x * x * x;
  float t = tanhf(0.7978845608f * (x + 0.044715f * x3));
  return 0.5f * x * (1.0f + t);
}

// CDNA5 async copy: global -> LDS, 16B per lane, tracked by ASYNCcnt.
__device__ inline void async_copy_b128(unsigned lds_byte_addr, const void* gaddr) {
  asm volatile("global_load_async_to_lds_b128 %0, %1, off"
               :: "v"(lds_byte_addr), "v"(gaddr)
               : "memory");
}
__device__ inline void wait_asynccnt0() {
  asm volatile("s_wait_asynccnt 0x0" ::: "memory");
}
// low 32 bits of a generic pointer to __shared__ = LDS byte offset (ISA 10.2)
__device__ inline unsigned lds_addr32(const void* p) {
  return (unsigned)(unsigned long long)p;
}

// A fragment (16x32 MxK bf16): lane<16 -> row=lane, K k0..k0+15
//                              lane>=16 -> row=lane-16, K k0+16..k0+31
__device__ inline v16bf loadA_frag(const unsigned short* base, int stride, int k0, int lane) {
  int r  = lane & 15;
  int kb = k0 + ((lane & 16) ? 16 : 0);
  const unsigned short* p = base + r * stride + kb;
  FragCvt t;
  t.u[0] = *reinterpret_cast<const U4*>(p);
  t.u[1] = *reinterpret_cast<const U4*>(p + 8);
  return t.v;
}

// B fragment (32x16 KxN bf16), weights stored [N,K] row-major:
// lane<16 -> col=lane, K k0..k0+15 ; lane>=16 -> col=lane-16, K k0+16..k0+31
__device__ inline v16bf loadB_frag(const unsigned short* base, int stride, int k0, int lane) {
  int n  = lane & 15;
  int kb = k0 + ((lane & 16) ? 16 : 0);
  const unsigned short* p = base + n * stride + kb;
  FragCvt t;
  t.u[0] = *reinterpret_cast<const U4*>(p);
  t.u[1] = *reinterpret_cast<const U4*>(p + 8);
  return t.v;
}

// ---------------- kernel 0: fp32 -> bf16 bulk convert ----------------
__global__ __launch_bounds__(256) void f2bf_kernel(const float* __restrict__ in,
                                                   unsigned short* __restrict__ out,
                                                   int n4) {
  int i = blockIdx.x * 256 + threadIdx.x;
  if (i >= n4) return;
  F4 v = reinterpret_cast<const F4*>(in)[i];
  union { unsigned short s[4]; unsigned long long u; } o;
  o.s[0] = f2bf(v.x); o.s[1] = f2bf(v.y); o.s[2] = f2bf(v.z); o.s[3] = f2bf(v.w);
  reinterpret_cast<unsigned long long*>(out)[i] = o.u;
}

// ---------------- kernel: init gate accumulator with gate_b2 ----------------
__global__ void init_g_kernel(const float* __restrict__ gb2, float* __restrict__ g) {
  int i = threadIdx.x;          // 64 threads: b*8+e
  if (i < Bn * En) g[i] = gb2[i % En];
}

// ---------------- gate kernel ----------------
// One block = one (b, 16-row s-tile). h = relu(x*gw1^T + gb1); partial
// g[b,e] += (1/S) * sum_{s,o} h[s,o]*gw2[e,o] via wave reduction + atomics.
__global__ __launch_bounds__(256) void gate_kernel(const unsigned short* __restrict__ xbf,
                                                   const unsigned short* __restrict__ gw1bf,
                                                   const float* __restrict__ gb1,
                                                   const float* __restrict__ gw2,
                                                   float* __restrict__ g) {
  constexpr int LSTR = Hn + 8;               // padded LDS row stride (halves)
  __shared__ unsigned short sX[16 * LSTR];

  const int tid  = threadIdx.x;
  const int lane = tid & 31;
  const int wid  = tid >> 5;
  const int blk  = blockIdx.x;
  const int b    = blk / (Sn / 16);
  const int s0   = (blk % (Sn / 16)) * 16;

  // stage 16 x 1024 bf16 A-tile into LDS via async copies
  const unsigned short* xsrc = xbf + ((size_t)b * Sn + s0) * Hn;
  #pragma unroll
  for (int i = 0; i < 8; ++i) {
    int chunk = tid + i * 256;              // 2048 chunks of 16B
    int row   = chunk >> 7;
    int col8  = (chunk & 127) * 8;
    async_copy_b128(lds_addr32(&sX[row * LSTR + col8]),
                    xsrc + (size_t)row * Hn + col8);
  }
  wait_asynccnt0();
  __syncthreads();

  float esum[En];
  #pragma unroll
  for (int e = 0; e < En; ++e) esum[e] = 0.f;

  for (int ot = wid; ot < Hn / 16; ot += 8) {
    const int o0 = ot * 16;
    v8f acc;
    #pragma unroll
    for (int i = 0; i < 8; ++i) acc[i] = 0.f;

    const int nB = lane & 15;
    const int kbB = (lane & 16) ? 16 : 0;
    const unsigned short* brow = gw1bf + (size_t)(o0 + nB) * Hn + kbB;

    for (int kt = 0; kt < Hn; kt += 32) {
      v16bf aF = loadA_frag(sX, LSTR, kt, lane);
      FragCvt t;
      t.u[0] = *reinterpret_cast<const U4*>(brow + kt);
      t.u[1] = *reinterpret_cast<const U4*>(brow + kt + 8);
      acc = __builtin_amdgcn_wmma_f32_16x16x32_bf16(false, aF, false, t.v,
                                                    (short)0, acc, false, false);
    }

    const int ocol = o0 + (lane & 15);
    const float b1 = gb1[ocol];
    float cs = 0.f;
    #pragma unroll
    for (int r = 0; r < 8; ++r) cs += fmaxf(acc[r] + b1, 0.f);
    #pragma unroll
    for (int e = 0; e < En; ++e) esum[e] += cs * gw2[e * Hn + ocol];
  }

  #pragma unroll
  for (int e = 0; e < En; ++e) {
    float v = esum[e];
    for (int off = 16; off > 0; off >>= 1) v += __shfl_xor(v, off, 32);
    if (lane == 0) atomicAdd(&g[b * En + e], v * (1.0f / Sn));
  }
}

// ---------------- merged bias: out[b,n] = base[n] + sum_e g[b,e]*task[e,n] ----------------
__global__ __launch_bounds__(256) void bias_merge_kernel(const float* __restrict__ base,
                                                         const float* __restrict__ task,
                                                         const float* __restrict__ g,
                                                         float* __restrict__ out, int N) {
  int i = blockIdx.x * 256 + threadIdx.x;
  if (i >= Bn * N) return;
  int b = i / N, n = i % N;
  float v = base[n];
  #pragma unroll
  for (int e = 0; e < En; ++e) v += g[b * En + e] * task[(size_t)e * N + n];
  out[i] = v;
}

// ---------------- merged weights (bf16): out[b,:] = base + sum_e g[b,e]*task[e,:] ----------------
__global__ __launch_bounds__(256) void weight_merge_kernel(const float* __restrict__ base,
                                                           const float* __restrict__ task,
                                                           const float* __restrict__ g,
                                                           unsigned short* __restrict__ out,
                                                           int NK) {
  __shared__ float sg[Bn * En];
  if (threadIdx.x < Bn * En) sg[threadIdx.x] = g[threadIdx.x];
  __syncthreads();

  int i4 = (blockIdx.x * 256 + threadIdx.x) * 4;
  if (i4 >= NK) return;

  F4 bs = *reinterpret_cast<const F4*>(base + i4);
  float acc[Bn][4];
  #pragma unroll
  for (int b = 0; b < Bn; ++b) { acc[b][0]=bs.x; acc[b][1]=bs.y; acc[b][2]=bs.z; acc[b][3]=bs.w; }

  #pragma unroll
  for (int e = 0; e < En; ++e) {
    const float* tp = task + (size_t)e * NK + i4;
    if (e + 1 < En) __builtin_prefetch(tp + NK, 0, 1);   // global_prefetch_b8
    F4 t = *reinterpret_cast<const F4*>(tp);
    #pragma unroll
    for (int b = 0; b < Bn; ++b) {
      float ge = sg[b * En + e];
      acc[b][0] += ge * t.x; acc[b][1] += ge * t.y;
      acc[b][2] += ge * t.z; acc[b][3] += ge * t.w;
    }
  }
  #pragma unroll
  for (int b = 0; b < Bn; ++b) {
    union { unsigned short s[4]; unsigned long long u; } o;
    #pragma unroll
    for (int j = 0; j < 4; ++j) o.s[j] = f2bf(acc[b][j]);
    *reinterpret_cast<unsigned long long*>(out + (size_t)b * NK + i4) = o.u;
  }
}

// ---------------- batched GEMM: C[z] = A[z] (MxK) * Bw[z] (NxK)^T + bias[z] ----------------
// Block tile 128(M) x 256(N), 8 waves as 2x4, wave tile 64x64 (16 WMMA tiles,
// each A/B fragment reused 4x -> 0.5 ds_load per WMMA). Double-buffered LDS
// filled with GLOBAL_LOAD_ASYNC_TO_LDS_B128 (ASYNCcnt); 32 WMMAs per K-step
// cover async latency. ACT==1: gelu + bf16 store (u). ACT==0: fp32 store.
template <int ACT>
__global__ __launch_bounds__(256) void gemm_kernel(const unsigned short* __restrict__ A,
                                                   const unsigned short* __restrict__ Bw,
                                                   const float* __restrict__ bias,
                                                   void* __restrict__ Cout,
                                                   int M, int N, int K) {
  constexpr int BK   = 64;        // K tile
  constexpr int TSTR = BK + 8;    // padded LDS row stride (halves); 144B, 16B-aligned
  __shared__ unsigned short sA[2][128 * TSTR];
  __shared__ unsigned short sB[2][256 * TSTR];

  const int tid  = threadIdx.x;
  const int lane = tid & 31;
  const int wid  = tid >> 5;
  const int wm   = wid & 1;    // 2 waves along M: 64 rows each
  const int wn   = wid >> 1;   // 4 waves along N: 64 cols each
  const int z    = blockIdx.z;

  const unsigned short* Ag = A  + (size_t)z * M * K + (size_t)(blockIdx.y * 128) * K;
  const unsigned short* Bg = Bw + (size_t)z * N * K + (size_t)(blockIdx.x * 256) * K;

  auto stage = [&](int buf, int kt) {
    #pragma unroll
    for (int i = 0; i < 4; ++i) {            // A: 128 rows x 128B = 1024 chunks
      int c = tid + i * 256;
      int row = c >> 3, col8 = (c & 7) * 8;
      async_copy_b128(lds_addr32(&sA[buf][row * TSTR + col8]),
                      Ag + (size_t)row * K + kt + col8);
    }
    #pragma unroll
    for (int i = 0; i < 8; ++i) {            // B: 256 rows x 128B = 2048 chunks
      int c = tid + i * 256;
      int row = c >> 3, col8 = (c & 7) * 8;
      async_copy_b128(lds_addr32(&sB[buf][row * TSTR + col8]),
                      Bg + (size_t)row * K + kt + col8);
    }
  };

  v8f acc[4][4];
  #pragma unroll
  for (int mt = 0; mt < 4; ++mt)
    #pragma unroll
    for (int nt = 0; nt < 4; ++nt)
      #pragma unroll
      for (int i = 0; i < 8; ++i) acc[mt][nt][i] = 0.f;

  // prologue: fill buffer 0
  stage(0, 0);
  wait_asynccnt0();
  __syncthreads();

  int cur = 0;
  for (int kt = 0; kt < K; kt += BK) {
    const int nxt = cur ^ 1;
    const bool more = (kt + BK) < K;
    if (more) stage(nxt, kt + BK);           // async fill of next buffer

    #pragma unroll
    for (int ks = 0; ks < BK; ks += 32) {
      v16bf aF[4], bF[4];
      #pragma unroll
      for (int mt = 0; mt < 4; ++mt)
        aF[mt] = loadA_frag(&sA[cur][(wm * 64 + mt * 16) * TSTR], TSTR, ks, lane);
      #pragma unroll
      for (int nt = 0; nt < 4; ++nt)
        bF[nt] = loadB_frag(&sB[cur][(wn * 64 + nt * 16) * TSTR], TSTR, ks, lane);
      #pragma unroll
      for (int mt = 0; mt < 4; ++mt)
        #pragma unroll
        for (int nt = 0; nt < 4; ++nt)
          acc[mt][nt] = __builtin_amdgcn_wmma_f32_16x16x32_bf16(
              false, aF[mt], false, bF[nt], (short)0, acc[mt][nt], false, false);
    }

    if (more) wait_asynccnt0();              // next buffer landed
    __syncthreads();                         // everyone done reading cur / sees nxt
    cur = nxt;
  }

  // epilogue
  const int rowbase = blockIdx.y * 128 + wm * 64;
  const int colbase = blockIdx.x * 256 + wn * 64;
  #pragma unroll
  for (int mt = 0; mt < 4; ++mt) {
    #pragma unroll
    for (int nt = 0; nt < 4; ++nt) {
      const int ncol = colbase + nt * 16 + (lane & 15);
      const float bv = bias[(size_t)z * N + ncol];
      #pragma unroll
      for (int r = 0; r < 8; ++r) {
        const int mrow = rowbase + mt * 16 + r + ((lane & 16) ? 8 : 0);
        float v = acc[mt][nt][r] + bv;
        if (ACT == 1) {
          v = gelu_tanh(v);
          reinterpret_cast<unsigned short*>(Cout)[(size_t)z * M * N + (size_t)mrow * N + ncol] = f2bf(v);
        } else {
          reinterpret_cast<float*>(Cout)[(size_t)z * M * N + (size_t)mrow * N + ncol] = v;
        }
      }
    }
  }
}

// ---------------- host side ----------------
extern "C" void kernel_launch(void* const* d_in, const int* in_sizes, int n_in,
                              void* d_out, int out_size, void* d_ws, size_t ws_size,
                              hipStream_t stream) {
  const float* x    = (const float*)d_in[0];
  const float* gw1  = (const float*)d_in[1];
  const float* gb1  = (const float*)d_in[2];
  const float* gw2  = (const float*)d_in[3];
  const float* gb2  = (const float*)d_in[4];
  const float* W1   = (const float*)d_in[5];
  const float* b1   = (const float*)d_in[6];
  const float* W2   = (const float*)d_in[7];
  const float* b2   = (const float*)d_in[8];
  const float* TW1  = (const float*)d_in[9];
  const float* Tb1  = (const float*)d_in[10];
  const float* TW2  = (const float*)d_in[11];
  const float* Tb2  = (const float*)d_in[12];

  char* w = (char*)d_ws;
  size_t off = 0;
  auto alloc = [&](size_t bytes) -> void* {
    void* p = w + off;
    off = (off + bytes + 255) & ~(size_t)255;
    return p;
  };
  unsigned short* xbf    = (unsigned short*)alloc((size_t)Bn * Sn * Hn * 2);
  unsigned short* gw1bf  = (unsigned short*)alloc((size_t)Hn * Hn * 2);
  float*          g      = (float*)alloc(Bn * En * 4);
  float*          mb1    = (float*)alloc((size_t)Bn * Fn * 4);
  float*          mb2    = (float*)alloc((size_t)Bn * Hn * 4);
  unsigned short* mW1bf  = (unsigned short*)alloc((size_t)Bn * Fn * Hn * 2);
  unsigned short* mW2bf  = (unsigned short*)alloc((size_t)Bn * Hn * Fn * 2);
  unsigned short* ubf    = (unsigned short*)alloc((size_t)Bn * Sn * Fn * 2);
  (void)ws_size; (void)in_sizes; (void)n_in; (void)out_size;

  // 0) convert x and gate_w1 to bf16; init g = gate_b2
  {
    int n4 = Bn * Sn * Hn / 4;
    f2bf_kernel<<<(n4 + 255) / 256, 256, 0, stream>>>(x, xbf, n4);
    int m4 = Hn * Hn / 4;
    f2bf_kernel<<<(m4 + 255) / 256, 256, 0, stream>>>(gw1, gw1bf, m4);
    init_g_kernel<<<1, 64, 0, stream>>>(gb2, g);
  }
  // 1) gate -> g[B,E]
  gate_kernel<<<Bn * Sn / 16, 256, 0, stream>>>(xbf, gw1bf, gb1, gw2, g);
  // 2) merged biases and weights
  bias_merge_kernel<<<(Bn * Fn + 255) / 256, 256, 0, stream>>>(b1, Tb1, g, mb1, Fn);
  bias_merge_kernel<<<(Bn * Hn + 255) / 256, 256, 0, stream>>>(b2, Tb2, g, mb2, Hn);
  weight_merge_kernel<<<(Fn * Hn / 4 + 255) / 256, 256, 0, stream>>>(W1, TW1, g, mW1bf, Fn * Hn);
  weight_merge_kernel<<<(Hn * Fn / 4 + 255) / 256, 256, 0, stream>>>(W2, TW2, g, mW2bf, Hn * Fn);
  // 3) u = gelu(x @ mW1^T + mb1)   [per-sample GEMM, bf16 out]
  gemm_kernel<1><<<dim3(Fn / 256, Sn / 128, Bn), 256, 0, stream>>>(
      xbf, mW1bf, mb1, ubf, Sn, Fn, Hn);
  // 4) out = u @ mW2^T + mb2       [fp32 out]
  gemm_kernel<0><<<dim3(Hn / 256, Sn / 128, Bn), 256, 0, stream>>>(
      ubf, mW2bf, mb2, d_out, Sn, Hn, Fn);
}
